// SDBGLAttention_3521873182868
// MI455X (gfx1250) — matrix operand order repaired and checked by reading, not compiled, not run on hardware
//
#include <hip/hip_runtime.h>
#include <hip/hip_bf16.h>
#include <math.h>

// ---------------------------------------------------------------------------
// SDBGLA spiking attention for MI455X (gfx1250, wave32, WMMA 16x16x32 f16)
// Spikes are binary -> exact in f16. Heavy GEMMs: v_wmma_f32_16x16x32_f16,
// 128x128 block tiles, async global->LDS staging (ASYNCcnt path).
// ---------------------------------------------------------------------------

typedef __attribute__((ext_vector_type(16))) _Float16 v16h;
typedef __attribute__((ext_vector_type(8)))  float    v8f;
typedef int v4i_ __attribute__((vector_size(16)));

static constexpr int T_ = 4;
static constexpr int B_ = 8;
static constexpr int N_ = 1024;
static constexpr int C_ = 512;
static constexpr int H_ = 8;
static constexpr int D_ = 64;
static constexpr float EPS_ = 1e-5f;

// ------------------------- async global->LDS helpers -----------------------
#if defined(__has_builtin)
#  if __has_builtin(__builtin_amdgcn_global_load_async_to_lds_b128)
#    define HAVE_ASYNC_B128 1
#  else
#    define HAVE_ASYNC_B128 0
#  endif
#  if __has_builtin(__builtin_amdgcn_s_wait_asynccnt)
#    define HAVE_WAIT_ASYNC 1
#  else
#    define HAVE_WAIT_ASYNC 0
#  endif
#else
#  define HAVE_ASYNC_B128 0
#  define HAVE_WAIT_ASYNC 0
#endif

__device__ __forceinline__ void cp_async_b128(const _Float16* g, _Float16* l) {
#if HAVE_ASYNC_B128
  // builtin signature (from hipcc diagnostic): arg0 = v4i* in global AS (non-const)
  __builtin_amdgcn_global_load_async_to_lds_b128(
      (__attribute__((address_space(1))) v4i_*)const_cast<_Float16*>(g),
      (__attribute__((address_space(3))) v4i_*)l, 0, 0);
#else
  unsigned lofs =
      (unsigned)(unsigned long long)(__attribute__((address_space(3))) void*)l;
  asm volatile("global_load_async_to_lds_b128 %0, %1, off"
               : : "v"(lofs), "v"(g) : "memory");
#endif
}

__device__ __forceinline__ void wait_async_all() {
#if HAVE_WAIT_ASYNC
  __builtin_amdgcn_s_wait_asynccnt(0);
#else
  asm volatile("s_wait_asynccnt 0x0" : : : "memory");
#endif
}

// ------------------------------- utility kernels ---------------------------

// f32 [C,C] -> f16 transposed [n][k] so GEMM B-tiles are contiguous row chunks.
__global__ void cvt_transpose_f16_kernel(const float* __restrict__ in,
                                         _Float16* __restrict__ out) {
  int i = blockIdx.x * blockDim.x + threadIdx.x;
  if (i >= C_ * C_) return;
  int n = i >> 9, k = i & (C_ - 1);
  out[i] = (_Float16)in[k * C_ + n];
}

__global__ void zero_kernel(float* __restrict__ p, int n) {
  int i = blockIdx.x * blockDim.x + threadIdx.x;
  if (i < n) p[i] = 0.f;
}

// Plain LIF scan over T (decay*v + x, spike = H(v - vth), hard reset).
template <typename OUT>
__global__ void lif_kernel(const float* __restrict__ in, OUT* __restrict__ out,
                           float decay, float vth, int S) {
  int s = blockIdx.x * blockDim.x + threadIdx.x;
  if (s >= S) return;
  float v = 0.f;
  for (int t = 0; t < T_; ++t) {
    v = decay * v + in[(size_t)t * S + s];
    float sp = (v - vth >= 0.f) ? 1.f : 0.f;
    out[(size_t)t * S + s] = (OUT)sp;
    v *= (1.f - sp);
  }
}

// ------------------------------- WMMA GEMM ---------------------------------
// out[M,512] = A[M,512](f16) @ W^T (f16, pre-transposed [n][k]) + bias.
// 128x128 block tile, 256 threads = 8 waves in a 4(m) x 2(n) grid; each wave
// owns 32x64 (2x4 accumulator tiles, 8 WMMAs per K-step). K stepped by 32.
// Tiles staged with GLOBAL_LOAD_ASYNC_TO_LDS_B128.
__global__ __launch_bounds__(256)
void gemm_wmma_kernel(const _Float16* __restrict__ A, const _Float16* __restrict__ Wt,
                      const float* __restrict__ bias, float* __restrict__ out) {
  constexpr int K = C_;      // 512
  constexpr int NC = C_;     // 512
  __shared__ _Float16 As[128 * 40];   // [m][k], padded (80B rows, 16B aligned)
  __shared__ _Float16 Bs[128 * 40];   // [n][k]
  const int tid  = threadIdx.x;
  const int lane = tid & 31;
  const int w    = tid >> 5;
  const int wm   = w >> 1;      // 0..3 -> 32-row strip
  const int wn   = w & 1;       // 0..1 -> 64-col strip
  const int lh   = lane >> 4;   // lane half (0/1)
  const int ll   = lane & 15;
  const int m0   = blockIdx.y * 128;
  const int n0   = blockIdx.x * 128;

  v8f acc[2][4] = {};

  for (int k0 = 0; k0 < K; k0 += 32) {
    if (k0 + 32 < K)
      __builtin_prefetch(&A[(size_t)(m0 + (tid >> 1)) * K + k0 + 32], 0, 3);
    // stage A 128x32 and B 128x32, 512 chunks of 8 halfs each
#pragma unroll
    for (int i = 0; i < 2; ++i) {
      int ck = tid + i * 256;
      int r = ck >> 2, c8 = (ck & 3) * 8;
      cp_async_b128(&A[(size_t)(m0 + r) * K + k0 + c8], &As[r * 40 + c8]);
      cp_async_b128(&Wt[(size_t)(n0 + r) * K + k0 + c8], &Bs[r * 40 + c8]);
    }
    wait_async_all();
    __syncthreads();

    union { v16h h; unsigned u[8]; } a[2];
#pragma unroll
    for (int mt = 0; mt < 2; ++mt) {
      const int mrow = wm * 32 + mt * 16 + ll;
#pragma unroll
      for (int r = 0; r < 8; ++r) {            // A-frag: VGPR r <-> K pair
        int kk = 2 * r + 16 * lh;
        a[mt].u[r] = *(const unsigned*)&As[mrow * 40 + kk];
      }
    }
#pragma unroll
    for (int nt = 0; nt < 4; ++nt) {
      union { v16h h; unsigned u[8]; } b;
      const int nrow = wn * 64 + nt * 16 + ll;
#pragma unroll
      for (int r = 0; r < 8; ++r) {            // B-frag: K pairs 2r(,+16)
        int kk = 2 * r + 16 * lh;
        b.u[r] = *(const unsigned*)&Bs[nrow * 40 + kk];
      }
#pragma unroll
      for (int mt = 0; mt < 2; ++mt)
        acc[mt][nt] = __builtin_amdgcn_wmma_f32_16x16x32_f16(
            false, a[mt].h, false, b.h, (short)0, acc[mt][nt], false, false);
    }
    __syncthreads();
  }

#pragma unroll
  for (int mt = 0; mt < 2; ++mt)
#pragma unroll
    for (int nt = 0; nt < 4; ++nt)
#pragma unroll
      for (int j = 0; j < 8; ++j) {            // C layout: VGPR j -> M=j(+8)
        int m = m0 + wm * 32 + mt * 16 + j + 8 * lh;
        int n = n0 + wn * 64 + nt * 16 + ll;
        out[(size_t)m * NC + n] = acc[mt][nt][j] + bias[n];
      }
}

// --------------------------- per-channel batch stats -----------------------
__global__ __launch_bounds__(512)
void colstats_kernel(const float* __restrict__ x, float* __restrict__ sums,
                     float* __restrict__ sqs) {
  int c  = threadIdx.x;          // C_ = 512 columns
  int r0 = blockIdx.x * 128;
  float s = 0.f, q = 0.f;
  for (int r = 0; r < 128; ++r) {
    float v = x[(size_t)(r0 + r) * C_ + c];
    s += v; q += v * v;
  }
  atomicAdd(&sums[c], s);
  atomicAdd(&sqs[c], q);
}

// --------------------------- fused BN + LIF scan ---------------------------
template <typename OUT>
__global__ void bn_lif_kernel(const float* __restrict__ pre,
                              const float* __restrict__ sums,
                              const float* __restrict__ sqs,
                              const float* __restrict__ g,
                              const float* __restrict__ beta,
                              OUT* __restrict__ spk, float* __restrict__ upool,
                              float decay, float vth, float invM, float invN) {
  const int S = B_ * N_ * C_;
  int s = blockIdx.x * blockDim.x + threadIdx.x;
  if (s >= S) return;
  int c = s & (C_ - 1);
  int b = s / (N_ * C_);
  float mu  = sums[c] * invM;
  float var = sqs[c] * invM - mu * mu;
  float inv = rsqrtf(var + EPS_);
  float gc = g[c], bc = beta[c];
  float v = 0.f;
  for (int t = 0; t < T_; ++t) {
    float xn = gc * (pre[(size_t)t * S + s] - mu) * inv + bc;
    v = decay * v + xn;
    float sp = (v - vth >= 0.f) ? 1.f : 0.f;
    spk[(size_t)t * S + s] = (OUT)sp;
    if (upool) atomicAdd(&upool[(t * B_ + b) * C_ + c], v * invN);  // pre-reset membrane
    v *= (1.f - sp);
  }
}

// ------------------------ gate path (single workgroup) ---------------------
__global__ __launch_bounds__(256)
void gate_kernel(const float* __restrict__ upool, const float* __restrict__ Wd,
                 const float* __restrict__ bd, const float* __restrict__ gd,
                 const float* __restrict__ betad, const float* __restrict__ wgate,
                 float* __restrict__ ubg, float* __restrict__ gam, float delta) {
  __shared__ float mu[64], rstd[64];
  const int tid = threadIdx.x;
  {  // row r = (t*B+b)*H + h = t*64 + bh
    int r = tid;
    int t = r >> 6;
    int bh = r & 63;
    int b = bh >> 3, h = bh & 7;
    const float* urow = upool + ((t * B_ + b) * C_ + h * D_);
    for (int d = 0; d < 64; ++d) {
      float acc = bd[d];
      for (int j = 0; j < 64; ++j) acc += urow[j] * Wd[j * 64 + d];
      ubg[r * 64 + d] = acc;
    }
  }
  __syncthreads();
  if (tid < 64) {
    float s = 0.f, q = 0.f;
    for (int r = 0; r < 256; ++r) { float v = ubg[r * 64 + tid]; s += v; q += v * v; }
    float m = s * (1.f / 256.f);
    mu[tid] = m;
    rstd[tid] = rsqrtf(q * (1.f / 256.f) - m * m + EPS_);
  }
  __syncthreads();
  for (int cc = tid; cc < B_ * H_ * D_; cc += 256) {
    int d = cc & 63;
    int bh = cc >> 6;
    float dec = 1.f - 1.f / (1.f + __expf(-wgate[d]));
    float gdd = gd[d], btd = betad[d], m = mu[d], rs = rstd[d];
    float v = 0.f;
    for (int t = 0; t < T_; ++t) {
      int r = t * 64 + bh;
      float xn = gdd * (ubg[r * 64 + d] - m) * rs + btd;
      v = dec * v + xn;
      float sp = (v - 1.f >= 0.f) ? 1.f : 0.f;
      gam[r * 64 + d] = 1.f - (1.f - delta) * sp;   // gamma_t
      v *= (1.f - sp);
    }
  }
}

// ----------------------- decay-gated linear attention ----------------------
// One block per (b,h). S_t = diag(gamma_t) S_{t-1} + K^T V ; O = scale * Q S.
__global__ __launch_bounds__(128)
void attn_kernel(const _Float16* __restrict__ qs, const _Float16* __restrict__ ks,
                 const _Float16* __restrict__ vs, const float* __restrict__ gam,
                 float* __restrict__ o, float scale) {
  __shared__ float    Ssm[64 * 64];   // state, f32 [d][e]
  __shared__ _Float16 Sh[64 * 64];    // state, f16 e-major [e][d] (WMMA B)
  __shared__ _Float16 Kt[64 * 36];    // [d][n_local]
  __shared__ _Float16 Vt[64 * 36];    // [e][n_local]
  __shared__ float    gsh[64];
  const int bh = blockIdx.x;
  const int b = bh >> 3, h = bh & 7;
  const int tid = threadIdx.x;
  const int lane = tid & 31, w = tid >> 5;
  const int lh = lane >> 4, ll = lane & 15;

  for (int i = tid; i < 4096; i += 128) Ssm[i] = 0.f;
  __syncthreads();

  for (int t = 0; t < T_; ++t) {
    if (tid < 64) gsh[tid] = gam[(t * 64 + bh) * 64 + tid];

    // ---- K^T V over N (contraction length 1024, step 32) ----
    v8f acc[4] = {};
    for (int nb = 0; nb < N_ / 32; ++nb) {
      int n0 = nb * 32;
      // stage transposed tiles: vectorized global reads, strided LDS stores
#pragma unroll
      for (int i = 0; i < 2; ++i) {
        int ck = tid + i * 128;                 // 256 chunks of 8 halfs
        int nl = ck >> 3, d8 = (ck & 7) * 8;
        size_t gi = ((size_t)(t * B_ + b) * N_ + (n0 + nl)) * C_ + h * D_ + d8;
        uint4 k4 = *(const uint4*)&ks[gi];
        uint4 v4 = *(const uint4*)&vs[gi];
        const _Float16* kh = (const _Float16*)&k4;
        const _Float16* vh = (const _Float16*)&v4;
#pragma unroll
        for (int j = 0; j < 8; ++j) {
          Kt[(d8 + j) * 36 + nl] = kh[j];
          Vt[(d8 + j) * 36 + nl] = vh[j];
        }
      }
      __syncthreads();
      union { v16h h; unsigned u[8]; } a;
      const int drow = w * 16 + ll;
#pragma unroll
      for (int r = 0; r < 8; ++r) {
        int kk = 2 * r + 16 * lh;
        a.u[r] = *(const unsigned*)&Kt[drow * 36 + kk];
      }
#pragma unroll
      for (int et = 0; et < 4; ++et) {
        union { v16h h; unsigned u[8]; } bb;
        const int erow = et * 16 + ll;
#pragma unroll
        for (int r = 0; r < 8; ++r) {
          int kk = 2 * r + 16 * lh;
          bb.u[r] = *(const unsigned*)&Vt[erow * 36 + kk];
        }
        acc[et] = __builtin_amdgcn_wmma_f32_16x16x32_f16(
            false, a.h, false, bb.h, (short)0, acc[et], false, false);
      }
      __syncthreads();
    }

    // ---- S = diag(gamma) * S + K^T V ; mirror into f16 (e-major) ----
#pragma unroll
    for (int et = 0; et < 4; ++et)
#pragma unroll
      for (int j = 0; j < 8; ++j) {
        int d = w * 16 + j + 8 * lh;
        int e = et * 16 + ll;
        float sn = gsh[d] * Ssm[d * 64 + e] + acc[et][j];
        Ssm[d * 64 + e] = sn;
        Sh[e * 64 + d] = (_Float16)sn;
      }
    __syncthreads();

    // ---- O = scale * Q @ S  (A frags straight from global, d-contiguous) ----
    for (int nb = 0; nb < N_ / 64; ++nb) {
      const int nrow = nb * 64 + w * 16 + ll;
      const _Float16* qrow = qs + ((size_t)(t * B_ + b) * N_ + nrow) * C_ + h * D_;
      union { v16h h; unsigned u[8]; } a0, a1;
#pragma unroll
      for (int r = 0; r < 8; ++r) {
        int kk = 2 * r + 16 * lh;
        a0.u[r] = *(const unsigned*)&qrow[kk];
        a1.u[r] = *(const unsigned*)&qrow[32 + kk];
      }
#pragma unroll
      for (int et = 0; et < 4; ++et) {
        v8f oc = {};
        const int erow = et * 16 + ll;
        union { v16h h; unsigned u[8]; } b0, b1;
#pragma unroll
        for (int r = 0; r < 8; ++r) {
          int kk = 2 * r + 16 * lh;
          b0.u[r] = *(const unsigned*)&Sh[erow * 64 + kk];
          b1.u[r] = *(const unsigned*)&Sh[erow * 64 + 32 + kk];
        }
        oc = __builtin_amdgcn_wmma_f32_16x16x32_f16(
            false, a0.h, false, b0.h, (short)0, oc, false, false);
        oc = __builtin_amdgcn_wmma_f32_16x16x32_f16(
            false, a1.h, false, b1.h, (short)0, oc, false, false);
#pragma unroll
        for (int j = 0; j < 8; ++j) {
          int m = nb * 64 + w * 16 + j + 8 * lh;
          int e = et * 16 + ll;
          o[((size_t)(t * B_ + b) * N_ + m) * C_ + h * D_ + e] = scale * oc[j];
        }
      }
    }
    __syncthreads();
  }
}

// ------------------------------- host launcher -----------------------------

extern "C" void kernel_launch(void* const* d_in, const int* in_sizes, int n_in,
                              void* d_out, int out_size, void* d_ws, size_t ws_size,
                              hipStream_t stream) {
  (void)in_sizes; (void)n_in; (void)out_size; (void)ws_size;
  const float* x     = (const float*)d_in[0];
  const float* Wq    = (const float*)d_in[1];
  const float* bq    = (const float*)d_in[2];
  const float* gq    = (const float*)d_in[3];
  const float* betaq = (const float*)d_in[4];
  const float* Wk    = (const float*)d_in[5];
  const float* bk    = (const float*)d_in[6];
  const float* gk    = (const float*)d_in[7];
  const float* betak = (const float*)d_in[8];
  const float* Wv    = (const float*)d_in[9];
  const float* bv    = (const float*)d_in[10];
  const float* gv    = (const float*)d_in[11];
  const float* betav = (const float*)d_in[12];
  const float* Wp    = (const float*)d_in[13];
  const float* bp    = (const float*)d_in[14];
  const float* gp    = (const float*)d_in[15];
  const float* betap = (const float*)d_in[16];
  const float* Wd    = (const float*)d_in[17];
  const float* bd    = (const float*)d_in[18];
  const float* gd    = (const float*)d_in[19];
  const float* betad = (const float*)d_in[20];
  const float* wgate = (const float*)d_in[21];

  const size_t TBNC = (size_t)T_ * B_ * N_ * C_;   // 16,777,216
  const int    S    = B_ * N_ * C_;                // 4,194,304
  const int    M    = T_ * B_ * N_;                // 32,768
  const int    CW   = C_ * C_;                     // 262,144

  size_t off = 0;
  auto take = [&](size_t bytes) -> void* {
    void* p = (char*)d_ws + off;
    off += (bytes + 255) & ~(size_t)255;
    return p;
  };
  _Float16* xs    = (_Float16*)take(TBNC * 2);
  float*    pre   = (float*)   take(TBNC * 4);   // pre-BN scratch, reused as attn out
  _Float16* q_spk = (_Float16*)take(TBNC * 2);   // reused as attn-LIF spikes
  _Float16* k_spk = (_Float16*)take(TBNC * 2);
  _Float16* v_spk = (_Float16*)take(TBNC * 2);
  _Float16* w16q  = (_Float16*)take((size_t)CW * 2);   // transposed f16 weights
  _Float16* w16k  = (_Float16*)take((size_t)CW * 2);
  _Float16* w16v  = (_Float16*)take((size_t)CW * 2);
  _Float16* w16p  = (_Float16*)take((size_t)CW * 2);
  float*    upool = (float*)   take((size_t)T_ * B_ * C_ * 4);  // 16384
  float*    ubg   = (float*)   take((size_t)T_ * B_ * C_ * 4);  // gate GEMM scratch
  float*    gamg  = (float*)   take((size_t)T_ * B_ * C_ * 4);  // gamma [T,B,H,D]
  float*    sums  = (float*)   take(1024 * 4);                  // sums[512]+sqs[512]
  float*    sqs   = sums + 512;

  const dim3 gemm_grid(C_ / 128, M / 128);   // (4, 256)
  const int  TPB = 256;
  const int  elw_blocks = (S + TPB - 1) / TPB;

  // weight conversion (f32 -> f16, transposed) + input LIF
  cvt_transpose_f16_kernel<<<(CW + TPB - 1) / TPB, TPB, 0, stream>>>(Wq, w16q);
  cvt_transpose_f16_kernel<<<(CW + TPB - 1) / TPB, TPB, 0, stream>>>(Wk, w16k);
  cvt_transpose_f16_kernel<<<(CW + TPB - 1) / TPB, TPB, 0, stream>>>(Wv, w16v);
  cvt_transpose_f16_kernel<<<(CW + TPB - 1) / TPB, TPB, 0, stream>>>(Wp, w16p);
  lif_kernel<_Float16><<<elw_blocks, TPB, 0, stream>>>(x, xs, 0.5f, 1.0f, S);

  // Q projection -> BN -> LIF
  gemm_wmma_kernel<<<gemm_grid, 256, 0, stream>>>(xs, w16q, bq, pre);
  zero_kernel<<<4, TPB, 0, stream>>>(sums, 1024);
  colstats_kernel<<<M / 128, 512, 0, stream>>>(pre, sums, sqs);
  bn_lif_kernel<_Float16><<<elw_blocks, TPB, 0, stream>>>(
      pre, sums, sqs, gq, betaq, q_spk, nullptr, 0.5f, 1.0f, 1.f / M, 0.f);

  // K projection -> BN -> LIF (+ pooled membrane)
  gemm_wmma_kernel<<<gemm_grid, 256, 0, stream>>>(xs, w16k, bk, pre);
  zero_kernel<<<4, TPB, 0, stream>>>(sums, 1024);
  zero_kernel<<<64, TPB, 0, stream>>>(upool, T_ * B_ * C_);
  colstats_kernel<<<M / 128, 512, 0, stream>>>(pre, sums, sqs);
  bn_lif_kernel<_Float16><<<elw_blocks, TPB, 0, stream>>>(
      pre, sums, sqs, gk, betak, k_spk, upool, 0.5f, 1.0f, 1.f / M, 1.f / N_);

  // V projection -> BN -> LIF
  gemm_wmma_kernel<<<gemm_grid, 256, 0, stream>>>(xs, w16v, bv, pre);
  zero_kernel<<<4, TPB, 0, stream>>>(sums, 1024);
  colstats_kernel<<<M / 128, 512, 0, stream>>>(pre, sums, sqs);
  bn_lif_kernel<_Float16><<<elw_blocks, TPB, 0, stream>>>(
      pre, sums, sqs, gv, betav, v_spk, nullptr, 0.5f, 1.0f, 1.f / M, 0.f);

  // gate path -> gamma
  gate_kernel<<<1, 256, 0, stream>>>(upool, Wd, bd, gd, betad, wgate, ubg, gamg, 0.5f);

  // decay-gated linear attention (o -> pre buffer)
  attn_kernel<<<B_ * H_, 128, 0, stream>>>(q_spk, k_spk, v_spk, gamg, pre, 0.125f);

  // attn LIF (threshold 0.5) -> spikes (reuse q_spk)
  lif_kernel<_Float16><<<elw_blocks, TPB, 0, stream>>>(pre, q_spk, 0.5f, 0.5f, S);

  // output projection -> BN -> LIF -> d_out (f32)
  gemm_wmma_kernel<<<gemm_grid, 256, 0, stream>>>(q_spk, w16p, bp, pre);
  zero_kernel<<<4, TPB, 0, stream>>>(sums, 1024);
  colstats_kernel<<<M / 128, 512, 0, stream>>>(pre, sums, sqs);
  bn_lif_kernel<float><<<elw_blocks, TPB, 0, stream>>>(
      pre, sums, sqs, gp, betap, (float*)d_out, nullptr, 0.5f, 1.0f, 1.f / M, 0.f);
}
